// StochasticMoELayer_27109833572674
// MI455X (gfx1250) — compile-verified, matrix operand order checked by reading, not run on hardware
//
#include <hip/hip_runtime.h>
#include <cstdint>
#include <cstddef>

// Problem constants (match reference)
#define E_ 8
#define T_ 8192
#define H_ 1024
#define I_ 4096

// Tiling
#define TM 16     // tokens per workgroup
#define KC 128    // intermediate-dim chunk

typedef float  v8f   __attribute__((ext_vector_type(8)));
typedef __bf16 v16bf __attribute__((ext_vector_type(16)));
typedef __bf16 bf16x8 __attribute__((ext_vector_type(8)));

__device__ __forceinline__ v8f wmma_bf16(v16bf a, v16bf b, v8f c) {
  // D(16x16 f32) = A(16x32 bf16) * B(32x16 bf16) + C
  return __builtin_amdgcn_wmma_f32_16x16x32_bf16(false, a, false, b, (short)0, c,
                                                 false, false);
}

// Fragment loader from row-major f32 memory (used for both A- and B-operands;
// the 16-bit A and B WMMA layouts are mirror images: lane%16 -> row (M or N),
// lane/16 -> K-half).  Covers rows [row0,row0+16) x K [k0,k0+32):
//   elems 0..7  <- K = k0 +      (lane/16)*8 + {0..7}
//   elems 8..15 <- K = k0 + 16 + (lane/16)*8 + {0..7}
__device__ __forceinline__ v16bf frag_f32_global(const float* __restrict__ base,
                                                 int ld, int row0, int k0, int lane) {
  const int g = lane >> 4, r = lane & 15;
  const float* p = base + (size_t)(row0 + r) * ld + (k0 + g * 8);
  const float4* q0 = (const float4*)p;
  const float4* q1 = (const float4*)(p + 16);
  float4 c0 = q0[0], c1 = q0[1];
  float4 c2 = q1[0], c3 = q1[1];
  v16bf f;
  f[0]  = (__bf16)c0.x; f[1]  = (__bf16)c0.y; f[2]  = (__bf16)c0.z; f[3]  = (__bf16)c0.w;
  f[4]  = (__bf16)c1.x; f[5]  = (__bf16)c1.y; f[6]  = (__bf16)c1.z; f[7]  = (__bf16)c1.w;
  f[8]  = (__bf16)c2.x; f[9]  = (__bf16)c2.y; f[10] = (__bf16)c2.z; f[11] = (__bf16)c2.w;
  f[12] = (__bf16)c3.x; f[13] = (__bf16)c3.y; f[14] = (__bf16)c3.z; f[15] = (__bf16)c3.w;
  return f;
}

// Same fragment layout, sourced from bf16 LDS.
__device__ __forceinline__ v16bf frag_bf16_lds(const __bf16* base, int ld,
                                               int row0, int k0, int lane) {
  const int g = lane >> 4, r = lane & 15;
  const __bf16* p = base + (row0 + r) * ld + (k0 + g * 8);
  bf16x8 c0 = *(const bf16x8*)p;
  bf16x8 c1 = *(const bf16x8*)(p + 16);
  v16bf f;
#pragma unroll
  for (int j = 0; j < 8; ++j) { f[j] = c0[j]; f[8 + j] = c1[j]; }
  return f;
}

// ---------------------------------------------------------------------------
// Kernel 0: zero the output (we accumulate per-expert contributions with
// f32 atomics; harness poisons d_out, so re-init every call).
// ---------------------------------------------------------------------------
__global__ void zero_out_kernel(float* __restrict__ out) {
  size_t i = ((size_t)blockIdx.x * blockDim.x + threadIdx.x) * 4;
  float4 z; z.x = 0.f; z.y = 0.f; z.z = 0.f; z.w = 0.f;
  *(float4*)(out + i) = z;
}

// ---------------------------------------------------------------------------
// Kernel 1: router.  One wave32 per token: 8 length-1024 dot products,
// butterfly reduce, softmax-equivalent top-2 with normalized weights,
// scatter into dense w[T][E] (zeros elsewhere).
// ---------------------------------------------------------------------------
__global__ void router_kernel(const float* __restrict__ x,
                              const float* __restrict__ rw,
                              float* __restrict__ wdense) {
  const int wave = threadIdx.x >> 5, lane = threadIdx.x & 31;
  const int t = blockIdx.x * 8 + wave;
  const float* xp = x + (size_t)t * H_;

  float acc[E_];
#pragma unroll
  for (int e = 0; e < E_; ++e) acc[e] = 0.f;

  for (int j = lane; j < H_; j += 32) {
    float xv = xp[j];
#pragma unroll
    for (int e = 0; e < E_; ++e) acc[e] += xv * rw[e * H_ + j];
  }
#pragma unroll
  for (int e = 0; e < E_; ++e) {
#pragma unroll
    for (int off = 16; off > 0; off >>= 1)
      acc[e] += __shfl_xor(acc[e], off, 32);
  }

  // top-2 on logits (softmax is monotonic; normalized pair weight is a
  // 2-way sigmoid of the logit gap).  Ties keep the lower index like top_k.
  int i1 = 0; float l1 = acc[0];
#pragma unroll
  for (int e = 1; e < E_; ++e) if (acc[e] > l1) { l1 = acc[e]; i1 = e; }
  int i2 = -1; float l2 = -3.4e38f;
#pragma unroll
  for (int e = 0; e < E_; ++e)
    if (e != i1 && acc[e] > l2) { l2 = acc[e]; i2 = e; }

  float w1 = 1.f / (1.f + __expf(l2 - l1));
  float w2 = 1.f - w1;
  if (lane < E_) {
    float wv = (lane == i1) ? w1 : ((lane == i2) ? w2 : 0.f);
    wdense[(size_t)t * E_ + lane] = wv;
  }
}

// ---------------------------------------------------------------------------
// Kernel 2: fused dense expert MLP.  grid = (T/TM, E).  Each workgroup:
//   - stages its 16-token x tile into LDS as bf16 (32 KB)
//   - loops intermediate dim in KC=128 chunks:
//       GEMM1: act = silu(x@Wg^T) * (x@Wu^T)  (wave w owns a 16-wide slice)
//       GEMM2: out_tile += act @ Wd^T          (wave w owns 128 out columns)
//   - scales by the routing weight and atomically adds into out.
// ---------------------------------------------------------------------------
__launch_bounds__(256)
__global__ void moe_expert_kernel(const float* __restrict__ x,
                                  const float* __restrict__ gup,
                                  const float* __restrict__ dwn,
                                  const float* __restrict__ wdense,
                                  float* __restrict__ out) {
  __shared__ __bf16 xs[TM * H_];    // 32 KB
  __shared__ __bf16 act[TM * KC];   // 4 KB
  __shared__ float  wtok[TM];

  const int e    = blockIdx.y;
  const int t0   = blockIdx.x * TM;
  const int tid  = threadIdx.x;
  const int wave = tid >> 5;
  const int lane = tid & 31;
  const int g    = lane >> 4;
  const int r    = lane & 15;

  const float* xT   = x   + (size_t)t0 * H_;
  const float* gupE = gup + (size_t)e * (2 * I_) * H_;
  const float* dwnE = dwn + (size_t)e * H_ * I_;

  // Stage x tile -> bf16 LDS (tile rows are contiguous in memory).
  for (int i = tid * 4; i < TM * H_; i += 256 * 4) {
    float4 v = *(const float4*)(xT + i);
    xs[i + 0] = (__bf16)v.x; xs[i + 1] = (__bf16)v.y;
    xs[i + 2] = (__bf16)v.z; xs[i + 3] = (__bf16)v.w;
  }
  if (tid < TM) wtok[tid] = wdense[(size_t)(t0 + tid) * E_ + e];
  __syncthreads();

  const v8f zz = {0.f, 0.f, 0.f, 0.f, 0.f, 0.f, 0.f, 0.f};
  v8f oacc[8];
#pragma unroll
  for (int nt = 0; nt < 8; ++nt) oacc[nt] = zz;

  for (int i0 = 0; i0 < I_; i0 += KC) {
    // ---- GEMM1: gate/up over K = H, wave w produces act cols [w*16, w*16+16)
    v8f gacc = zz, uacc = zz;
    const int ng = i0 + wave * 16;
#pragma unroll 4
    for (int k0 = 0; k0 < H_; k0 += 32) {
      v16bf a  = frag_bf16_lds(xs, H_, 0, k0, lane);
      v16bf bg = frag_f32_global(gupE, H_, ng, k0, lane);        // gate rows
      v16bf bu = frag_f32_global(gupE, H_, I_ + ng, k0, lane);   // up rows
      gacc = wmma_bf16(a, bg, gacc);
      uacc = wmma_bf16(a, bu, uacc);
    }
    // silu(gate)*up -> bf16 act tile (C-layout: lane%16 -> col, vgpr+8*(lane/16) -> row)
#pragma unroll
    for (int vv = 0; vv < 8; ++vv) {
      float gv = gacc[vv], uv = uacc[vv];
      float s  = gv / (1.f + __expf(-gv));
      int tok  = g * 8 + vv;
      act[tok * KC + wave * 16 + r] = (__bf16)(s * uv);
    }
    __syncthreads();

    // ---- GEMM2: out[0:16, wave*128 : wave*128+128) += act @ Wd^T
#pragma unroll
    for (int kk = 0; kk < KC; kk += 32) {
      v16bf a = frag_bf16_lds(act, KC, 0, kk, lane);
#pragma unroll
      for (int nt = 0; nt < 8; ++nt) {
        v16bf bd = frag_f32_global(dwnE, I_, wave * 128 + nt * 16, i0 + kk, lane);
        oacc[nt] = wmma_bf16(a, bd, oacc[nt]);
      }
    }
    __syncthreads();  // act is rewritten next chunk
  }

  // Scale by routing weight, accumulate into out (f32 atomic add).
#pragma unroll
  for (int nt = 0; nt < 8; ++nt) {
    const int col = wave * 128 + nt * 16 + r;
#pragma unroll
    for (int vv = 0; vv < 8; ++vv) {
      const int tok = g * 8 + vv;
      float contrib = wtok[tok] * oacc[nt][vv];
      __hip_atomic_fetch_add(&out[(size_t)(t0 + tok) * H_ + col], contrib,
                             __ATOMIC_RELAXED, __HIP_MEMORY_SCOPE_AGENT);
    }
  }
}

// ---------------------------------------------------------------------------
extern "C" void kernel_launch(void* const* d_in, const int* in_sizes, int n_in,
                              void* d_out, int out_size, void* d_ws, size_t ws_size,
                              hipStream_t stream) {
  (void)in_sizes; (void)n_in; (void)out_size; (void)ws_size;
  const float* x   = (const float*)d_in[0];  // [T, H]
  const float* rw  = (const float*)d_in[1];  // [E, H]
  const float* gup = (const float*)d_in[2];  // [E, 2I, H]
  const float* dwn = (const float*)d_in[3];  // [E, H, I]
  float* out    = (float*)d_out;             // [T, H]
  float* wdense = (float*)d_ws;              // [T, E] f32 (256 KB scratch)

  // out = 0
  zero_out_kernel<<<dim3((T_ * H_) / (256 * 4)), dim3(256), 0, stream>>>(out);
  // router -> dense per-(token,expert) combine weights
  router_kernel<<<dim3(T_ / 8), dim3(256), 0, stream>>>(x, rw, wdense);
  // dense fused expert MLP, weighted accumulate
  moe_expert_kernel<<<dim3(T_ / TM, E_), dim3(256), 0, stream>>>(x, gup, dwn, wdense, out);
}
